// Model_57217554317716
// MI455X (gfx1250) — matrix operand order repaired and checked by reading, not compiled, not run on hardware
//
#include <hip/hip_runtime.h>

#define DIMS   10000
#define KPAD   10016   // 313 * 32, zero-padded K for the WMMA loop
#define NPIX   784
#define NBATCH 32
#define NCLASS 10

typedef __attribute__((ext_vector_type(16))) _Float16 v16h;
typedef __attribute__((ext_vector_type(8)))  _Float16 v8h;
typedef __attribute__((ext_vector_type(8)))  float    v8f;

// ---------------------------------------------------------------------------
// Kernel 1: convert classify_weight [10][10000] f32 -> cw16 [16][KPAD] f16,
// zero-padded in both N (10->16) and K (10000->10016) so the WMMA loop
// accumulates exact zeros in the padding.
// ---------------------------------------------------------------------------
__global__ __launch_bounds__(256) void hd_prep_cw(const float* __restrict__ cw,
                                                  _Float16* __restrict__ cw16) {
    int i = blockIdx.x * 256 + threadIdx.x;
    if (i >= 16 * KPAD) return;
    int n = i / KPAD;
    int k = i - n * KPAD;
    float v = (n < NCLASS && k < DIMS) ? cw[n * DIMS + k] : 0.0f;
    cw16[i] = (_Float16)v;
}

// ---------------------------------------------------------------------------
// Kernel 2: encode.  hv[b,d] = sum_p pos[p,d] * level[idx[b,p], d];
// enc[b,d] = hv>0 ? +1 : -1 (f16).  One block = 64 consecutive d's x all 32
// batches.  idx table (32x784, u16) lives in LDS (50 KB); each lane handles a
// float2 d-pair for 4 batches -> 8 f32 accumulators.  level gathers are
// L2-resident (level_weight = 40 MB << 192 MB L2).
// ---------------------------------------------------------------------------
__global__ __launch_bounds__(256) void hd_encode(const float* __restrict__ x,
                                                 const float* __restrict__ pos,
                                                 const float* __restrict__ lev,
                                                 _Float16* __restrict__ enc) {
    __shared__ unsigned short sidx[NBATCH * NPIX];   // 50176 B

    int tid = threadIdx.x;
    for (int i = tid; i < NBATCH * NPIX; i += 256) {
        int q = (int)rintf(x[i] * 999.0f);           // round-half-even, matches jnp.round
        q = q < 0 ? 0 : (q > 999 ? 999 : q);
        sidx[i] = (unsigned short)q;
    }
    __syncthreads();

    int dlane = tid & 31;          // 32 d-pairs per wave
    int bgrp  = tid >> 5;          // 8 groups of 4 batches
    int d = blockIdx.x * 64 + dlane * 2;

    float ax[4] = {0.f, 0.f, 0.f, 0.f};
    float ay[4] = {0.f, 0.f, 0.f, 0.f};

    if (d < DIMS) {
        const float* posd = pos + d;
        const float* levd = lev + d;
        const unsigned short* idx0 = sidx + (bgrp * 4) * NPIX;
        for (int p = 0; p < NPIX; ++p) {
            __builtin_prefetch(posd + (p + 12) * DIMS, 0, 1);   // global_prefetch_b8
            float2 pv = *(const float2*)(posd + p * DIMS);
            #pragma unroll
            for (int j = 0; j < 4; ++j) {
                unsigned off = (unsigned)idx0[j * NPIX + p] * (unsigned)DIMS;
                float2 lv = *(const float2*)(levd + off);       // L2-resident gather
                ax[j] = fmaf(pv.x, lv.x, ax[j]);
                ay[j] = fmaf(pv.y, lv.y, ay[j]);
            }
        }
    }

    #pragma unroll
    for (int j = 0; j < 4; ++j) {
        int b = bgrp * 4 + j;
        _Float16* row = enc + (size_t)b * KPAD;
        if (d < DIMS) {
            row[d]     = ax[j] > 0.0f ? (_Float16)1.0f : (_Float16)-1.0f;
            row[d + 1] = ay[j] > 0.0f ? (_Float16)1.0f : (_Float16)-1.0f;
        } else {                                   // zero-fill K padding [10000,10016)
            if (d     < KPAD) row[d]     = (_Float16)0.0f;
            if (d + 1 < KPAD) row[d + 1] = (_Float16)0.0f;
        }
    }
}

// ---------------------------------------------------------------------------
// Kernel 3: classify GEMM via WMMA.  logits[32][10] = enc[32][K] @ cw16ᵀ.
// One block, 2 waves; wave w owns batches 16w..16w+15 (one 16x16 C tile).
// K loop: 313 x v_wmma_f32_16x16x32_f16 with f32 accumulate.
// A layout: lane l -> M=l&15, halves = K[kh..kh+7] ++ K[kh+8..kh+15],
// kh = (l>>4)*16 (two 16-B loads).  B layout: lane l -> N=l&15,
// halves = K[(l>>4)*16 .. +15] (one 32-B load).
// ---------------------------------------------------------------------------
__global__ __launch_bounds__(64) void hd_classify(const _Float16* __restrict__ enc,
                                                  const _Float16* __restrict__ cw16,
                                                  float* __restrict__ out) {
    int lane = threadIdx.x & 31;
    int wave = threadIdx.x >> 5;        // M-tile index
    int half = lane >> 4;
    int mrow = lane & 15;

    const _Float16* arow = enc  + (size_t)(wave * 16 + mrow) * KPAD + half * 16;
    const _Float16* brow = cw16 + (size_t)mrow * KPAD + half * 16;

    v8f c = {};
    for (int kb = 0; kb < KPAD; kb += 32) {
        union { v16h v; v8h h[2]; } a;
        a.h[0] = *(const v8h*)(arow + kb);
        a.h[1] = *(const v8h*)(arow + kb + 8);
        v16h b = *(const v16h*)(brow + kb);
        // D = A*B + C, f32 accumulate
        c = __builtin_amdgcn_wmma_f32_16x16x32_f16(
                /*neg_a=*/false, a.v, /*neg_b=*/false, b,
                /*c_mod=*/(short)0, c, /*reuse_a=*/false, /*reuse_b=*/false);
    }

    int n = mrow;                       // C: N = lane&15
    if (n < NCLASS) {
        #pragma unroll
        for (int r = 0; r < 8; ++r) {
            int M = r + half * 8;       // C: M = vgpr + (lane>=16)*8
            out[(wave * 16 + M) * NCLASS + n] = c[r];
        }
    }
}

// ---------------------------------------------------------------------------
extern "C" void kernel_launch(void* const* d_in, const int* in_sizes, int n_in,
                              void* d_out, int out_size, void* d_ws, size_t ws_size,
                              hipStream_t stream) {
    (void)in_sizes; (void)n_in; (void)out_size; (void)ws_size;

    const float* x   = (const float*)d_in[0];   // [32,1,28,28]
    const float* pos = (const float*)d_in[1];   // [784, 10000]
    const float* lev = (const float*)d_in[2];   // [1000, 10000]
    const float* cw  = (const float*)d_in[3];   // [10, 10000]

    char* ws = (char*)d_ws;
    _Float16* cw16 = (_Float16*)ws;                              // 16*KPAD f16 (313 KB)
    _Float16* enc  = (_Float16*)(ws + (size_t)16 * KPAD * 2);    // 32*KPAD f16 (626 KB)
    float* out = (float*)d_out;                                  // [32,10]

    hd_prep_cw<<<(16 * KPAD + 255) / 256, 256, 0, stream>>>(cw, cw16);
    hd_encode<<<(KPAD + 63) / 64, 256, 0, stream>>>(x, pos, lev, enc);
    hd_classify<<<1, 64, 0, stream>>>(enc, cw16, out);
}
